// STPT_9285719294257
// MI455X (gfx1250) — matrix operand order repaired and checked by reading, not compiled, not run on hardware
//
#include <hip/hip_runtime.h>

// ---------------------------------------------------------------------------
// Problem constants: B=8, T=2048, D=1024, H=2, DK=512
// ---------------------------------------------------------------------------
#define Bn     8
#define Tn     2048
#define Dn     1024
#define Hn     2
#define DKn    512
#define MROWS  (Bn * Tn)          // 16384

#define CC     32                 // chunk length (== WMMA K)
#define KBW    32                 // k-columns per workgroup
#define NKB    (DKn / KBW)        // 16
#define NCHUNK (Tn / CC)          // 64

typedef __bf16 v16bf __attribute__((ext_vector_type(16)));
typedef __bf16 v8bf  __attribute__((ext_vector_type(8)));
typedef float  v8f   __attribute__((ext_vector_type(8)));
typedef float  vf4   __attribute__((ext_vector_type(4)));

// ---------------------------------------------------------------------------
// WMMA fragment helpers (CDNA5 16x16x32 bf16, wave32; ISA 7.12.2 layouts).
// All operand tiles are staged so each lane's data is CONTIGUOUS in LDS:
//   A  (16x32): lane m=l%16; lo lanes K={0..7,16..23}, hi K={8..15,24..31}
//   B  (32x16): lane n=l%16; lo lanes K={0..7,16..23}, hi K={8..15,24..31} (N-major tile)
//   C/D(16x16): lane n=l%16; VGPR i -> row (i + (l<16?0:8)).
// ---------------------------------------------------------------------------
__device__ __forceinline__ int lane_id() { return (int)(threadIdx.x & 31); }

__device__ __forceinline__ v16bf cat8(v8bf lo, v8bf hi) {
    return __builtin_shufflevector(lo, hi, 0,1,2,3,4,5,6,7,8,9,10,11,12,13,14,15);
}

// A fragment from row-major MxK bf16 tile (rows 16B-aligned, ld % 8 == 0)
__device__ __forceinline__ v16bf load_a16x32_bf(const __bf16* p, int ld) {
    int l = lane_id();
    const __bf16* row = p + (l & 15) * ld;
    int kb = (l < 16) ? 0 : 8;
    v8bf lo = *(const v8bf*)(row + kb);
    v8bf hi = *(const v8bf*)(row + kb + 16);
    return cat8(lo, hi);
}

// B fragment from N-MAJOR (NxK) bf16 tile: lane n reads row n, K contiguous
__device__ __forceinline__ v16bf load_b32x16_nmajor_bf(const __bf16* p, int ld) {
    int l = lane_id();
    const __bf16* row = p + (l & 15) * ld;
    int kb = (l < 16) ? 0 : 8;
    v8bf lo = *(const v8bf*)(row + kb);
    v8bf hi = *(const v8bf*)(row + kb + 16);
    return cat8(lo, hi);
}

// Masked A from row-major f32 tile (causal, inclusive diagonal): k<=m ? P : 0
__device__ __forceinline__ v16bf load_a16x32_masked_f32(const float* p, int ld,
                                                        int mbase) {
    int l = lane_id();
    int m = l & 15;
    const float* row = p + m * ld;
    int mg = mbase + m;
    int kb = (l < 16) ? 0 : 8;
    vf4 f0 = *(const vf4*)(row + kb);
    vf4 f1 = *(const vf4*)(row + kb + 4);
    vf4 f2 = *(const vf4*)(row + kb + 16);
    vf4 f3 = *(const vf4*)(row + kb + 20);
    v16bf a;
#pragma unroll
    for (int j = 0; j < 4; ++j) {
        a[j]      = (__bf16)((kb + j      <= mg) ? f0[j] : 0.0f);
        a[4 + j]  = (__bf16)((kb + 4 + j  <= mg) ? f1[j] : 0.0f);
        a[8 + j]  = (__bf16)((kb + 16 + j <= mg) ? f2[j] : 0.0f);
        a[12 + j] = (__bf16)((kb + 20 + j <= mg) ? f3[j] : 0.0f);
    }
    return a;
}

__device__ __forceinline__ v8f load_c16x16_f32(const float* p, int ld) {
    int l = lane_id();
    int n = l & 15;
    int mb = (l < 16) ? 0 : 8;
    v8f c;
#pragma unroll
    for (int i = 0; i < 8; ++i) c[i] = p[(mb + i) * ld + n];
    return c;
}

__device__ __forceinline__ void store_c16x16_f32(float* p, int ld, v8f c) {
    int l = lane_id();
    int n = l & 15;
    int mb = (l < 16) ? 0 : 8;
#pragma unroll
    for (int i = 0; i < 8; ++i) p[(mb + i) * ld + n] = c[i];
}

__device__ __forceinline__ v8f wmma_bf16(v16bf a, v16bf b, v8f c) {
    return __builtin_amdgcn_wmma_f32_16x16x32_bf16(false, a, false, b,
                                                   (short)0, c, false, false);
}

// ---------------------------------------------------------------------------
// Kernel 1/4: Y = X @ W^T (M=16384, N=K=1024), bf16 WMMA, f32 accumulate.
// MODE is compile-time so each epilogue is straight-line code:
//   MODE 0: f32 output, standard (B,T,D) layout
//   MODE 1: f32 output, head-major (B,H,T,DK)
//   MODE 2: bf16 output, head-major (B,H,T,DK)   <- q and v for the scan
// ---------------------------------------------------------------------------
template <int MODE>
__global__ __launch_bounds__(256) void gemm_xwT_kernel(
    const float* __restrict__ X, const float* __restrict__ W,
    void* __restrict__ Y) {
    __shared__ __attribute__((aligned(16))) __bf16 As[128][40]; // 128x32 tile
    __shared__ __attribute__((aligned(16))) __bf16 Bs[64][40];  // Bs[n][k]=W[n][k]

    const int m0 = blockIdx.x * 128;
    const int n0 = blockIdx.y * 64;
    const int tid = threadIdx.x;
    const int wave = tid >> 5;
    const int lane = tid & 31;

    v8f acc[4];
#pragma unroll
    for (int ct = 0; ct < 4; ++ct)
#pragma unroll
        for (int i = 0; i < 8; ++i) acc[ct][i] = 0.0f;

    for (int k0 = 0; k0 < Dn; k0 += 32) {
        {   // stage X tile (128x32) via 16B loads/stores
            int r = tid >> 1, c0 = (tid & 1) * 16;
            const float* src = X + (size_t)(m0 + r) * Dn + k0 + c0;
#pragma unroll
            for (int c = 0; c < 16; c += 8) {
                vf4 x0 = *(const vf4*)(src + c);
                vf4 x1 = *(const vf4*)(src + c + 4);
                v8bf pk;
#pragma unroll
                for (int j = 0; j < 4; ++j) { pk[j] = (__bf16)x0[j]; pk[4+j] = (__bf16)x1[j]; }
                *(v8bf*)(&As[r][c0 + c]) = pk;
            }
        }
        {   // stage W tile N-major (straight copy)
            int r = tid >> 2, c0 = (tid & 3) * 8;
            const float* src = W + (size_t)(n0 + r) * Dn + k0 + c0;
            vf4 w0 = *(const vf4*)(src);
            vf4 w1 = *(const vf4*)(src + 4);
            v8bf pk;
#pragma unroll
            for (int j = 0; j < 4; ++j) { pk[j] = (__bf16)w0[j]; pk[4+j] = (__bf16)w1[j]; }
            *(v8bf*)(&Bs[r][c0]) = pk;
        }
        __syncthreads();

        v16bf a = load_a16x32_bf(&As[wave * 16][0], 40);
#pragma unroll
        for (int ct = 0; ct < 4; ++ct) {
            v16bf b = load_b32x16_nmajor_bf(&Bs[ct * 16][0], 40);
            acc[ct] = wmma_bf16(a, b, acc[ct]);
        }
        __syncthreads();
    }

    const int nl = lane & 15;
    const int mb = (lane < 16) ? 0 : 8;
#pragma unroll
    for (int ct = 0; ct < 4; ++ct) {
#pragma unroll
        for (int i = 0; i < 8; ++i) {
            int m = m0 + wave * 16 + mb + i;
            int n = n0 + ct * 16 + nl;
            float val = acc[ct][i];
            if constexpr (MODE == 0) {
                ((float*)Y)[(size_t)m * Dn + n] = val;
            } else {
                int b = m >> 11, t = m & (Tn - 1);
                int h = n >> 9, d = n & (DKn - 1);
                size_t idx = (size_t)((b * Hn + h) * Tn + t) * DKn + d;
                if constexpr (MODE == 1) ((float*)Y)[idx] = val;
                else                     ((__bf16*)Y)[idx] = (__bf16)val;
            }
        }
    }
}

// ---------------------------------------------------------------------------
// Kernel 2/4: normalize each k row over DK=512
// ---------------------------------------------------------------------------
__global__ __launch_bounds__(256) void knorm_kernel(float* __restrict__ kbuf) {
    __shared__ float red[256];
    float* p = kbuf + (size_t)blockIdx.x * DKn;
    int tid = threadIdx.x;
    float a = p[tid], b = p[tid + 256];
    red[tid] = a * a + b * b;
    __syncthreads();
#pragma unroll
    for (int off = 128; off > 0; off >>= 1) {
        if (tid < off) red[tid] += red[tid + off];
        __syncthreads();
    }
    float inv = 1.0f / fmaxf(sqrtf(red[0]), 1e-12f);
    p[tid] = a * inv;
    p[tid + 256] = b * inv;
}

// ---------------------------------------------------------------------------
// Kernel 3/4: chunked GLA scan. WG per (b,h,32-column block). State kept
// transposed: St[kcol][dv] (f32 master) + Stb (bf16 shadow for the Q@S GEMM).
// Per chunk: P=Qc Vc^T ; Yinter=Qc S0 ; y=G.*(Yinter+(P.*mask)K2)
//            St <- Gend .* (St + K2^T Vc)
// ---------------------------------------------------------------------------
#define ST_LD   520   // f32 state rows (2080B, 16B mult)
#define STB_LD  520   // bf16 shadow rows (1040B)
#define QS_LD   520   // bf16
#define VST_LD  40    // bf16
#define K2T_LD  40    // bf16
#define PS_LD   40    // f32
#define KS_LD   36    // f32
#define GS_LD   36
#define YS_LD   36

#define OFF_ST   0
#define OFF_STB  (OFF_ST  + KBW * ST_LD * 4)     //  66560
#define OFF_QS   (OFF_STB + KBW * STB_LD * 2)    // +33280
#define OFF_VS   (OFF_QS  + CC * QS_LD * 2)      // +33280
#define OFF_VST  (OFF_VS  + CC * QS_LD * 2)      // +33280
#define OFF_K2T  (OFF_VST + 512 * VST_LD * 2)    // +40960
#define OFF_PS   (OFF_K2T + CC * K2T_LD * 2)     // +2560
#define OFF_KS   (OFF_PS  + CC * PS_LD * 4)      // +5120
#define OFF_GS   (OFF_KS  + CC * KS_LD * 4)      // +4608
#define OFF_YS   (OFF_GS  + CC * GS_LD * 4)      // +4608
#define SCAN_LDS_BYTES (OFF_YS + CC * YS_LD * 4) // 228864 B (< 320KB WGP LDS)

__global__ __launch_bounds__(256) void gla_scan_kernel(
    const __bf16* __restrict__ qg, const float* __restrict__ kg,
    const __bf16* __restrict__ vg, float* __restrict__ yg,
    const float* __restrict__ Wgam, const float* __restrict__ bgam,
    const float* __restrict__ Wlam, const float* __restrict__ blam) {
    extern __shared__ __attribute__((aligned(16))) char smem[];
    float*  St  = (float*)(smem + OFF_ST);    // [KBW][ST_LD]  state^T f32
    __bf16* Stb = (__bf16*)(smem + OFF_STB);  // [KBW][STB_LD] state^T bf16
    __bf16* Qs  = (__bf16*)(smem + OFF_QS);   // [CC][QS_LD]   (t,dv)
    __bf16* Vs  = (__bf16*)(smem + OFF_VS);   // [CC][QS_LD]   (t,dv)
    __bf16* Vst = (__bf16*)(smem + OFF_VST);  // [512][VST_LD] (dv,t)
    __bf16* K2t = (__bf16*)(smem + OFF_K2T);  // [KBW][K2T_LD] (kcol,t)
    float*  Ps  = (float*)(smem + OFF_PS);    // [CC][PS_LD]   (t,t')
    float*  Ks  = (float*)(smem + OFF_KS);    // [CC][KS_LD]   (t,kcol)
    float*  Gs  = (float*)(smem + OFF_GS);    // [CC][GS_LD]   cumulative decay
    float*  Ys  = (float*)(smem + OFF_YS);    // [CC][YS_LD]   inter-chunk partial

    const int wg   = blockIdx.x;              // b*32 + h*16 + kb
    const int kb   = wg & (NKB - 1);
    const int h    = (wg >> 4) & (Hn - 1);
    const int b    = wg >> 5;
    const int tid  = threadIdx.x;
    const int wave = tid >> 5;
    const int lane = tid & 31;

    const size_t base = (size_t)(b * Hn + h) * Tn * DKn;
    const __bf16* qp = qg + base;
    const __bf16* vp = vg + base;
    const float*  kp = kg + base;

    for (int i = tid; i < KBW * ST_LD; i += 256) St[i] = 0.0f;
    for (int i = tid; i < KBW * STB_LD; i += 256) Stb[i] = (__bf16)0.0f;

    float wgp = 0.f, bgv = 0.f, wlp = 0.f, blv = 0.f;
    if (tid < KBW) {
        int dg = kb * KBW + tid;
        wgp = Wgam[h * DKn + dg];  bgv = bgam[h * DKn + dg];
        wlp = Wlam[h * DKn + dg];  blv = blam[h * DKn + dg];
    }
    __syncthreads();

    for (int ch = 0; ch < NCHUNK; ++ch) {
        const int t0 = ch * CC;
        {   // stage Q/V (bf16 in global): pure 16B copies + transposed V scatter
            int r = tid >> 3;
            int c0 = (tid & 7) * 64;
            const __bf16* qsrc = qp + (size_t)(t0 + r) * DKn + c0;
            const __bf16* vsrc = vp + (size_t)(t0 + r) * DKn + c0;
            for (int c = 0; c < 64; c += 8) {
                v8bf qv = *(const v8bf*)(qsrc + c);
                v8bf vv = *(const v8bf*)(vsrc + c);
                *(v8bf*)(Qs + r * QS_LD + c0 + c) = qv;
                *(v8bf*)(Vs + r * QS_LD + c0 + c) = vv;
#pragma unroll
                for (int j = 0; j < 8; ++j)
                    Vst[(c0 + c + j) * VST_LD + r] = vv[j];
            }
        }
        {   // stage this block's k columns (32x32 f32)
            int r = tid >> 3;
            int c0 = (tid & 7) * 4;
            const float* ksrc = kp + (size_t)(t0 + r) * DKn + kb * KBW + c0;
            *(vf4*)(Ks + r * KS_LD + c0) = *(const vf4*)ksrc;
        }
        __syncthreads();

        // gates + cumulative per-column decay (wave 0, one lane per column)
        if (tid < KBW) {
            float G = 1.0f;
            for (int t = 0; t < CC; ++t) {
                float kv  = Ks[t * KS_LD + tid];
                float gam = 1.0f / (1.0f + __expf(-(wgp * kv + bgv)));
                float lam = 1.0f / (1.0f + __expf(-(wlp * kv + blv)));
                float rho = (1.0f - lam) * gam;
                G *= rho;
                Gs[t * GS_LD + tid] = G;
                K2t[tid * K2T_LD + t] = (__bf16)((1.0f - rho) * kv / G);
            }
        }
        __syncthreads();

        if (wave < 4) {
            // GEMM1: P(32x32) = Qc @ Vc^T ; B rows are Vs[t'][dv] (N-major)
            int mt = (wave >> 1) & 1, nt = wave & 1;
            v8f c;
#pragma unroll
            for (int i = 0; i < 8; ++i) c[i] = 0.0f;
            for (int ks = 0; ks < DKn; ks += 32) {
                v16bf a  = load_a16x32_bf(Qs + mt * 16 * QS_LD + ks, QS_LD);
                v16bf bt = load_b32x16_nmajor_bf(Vs + nt * 16 * QS_LD + ks, QS_LD);
                c = wmma_bf16(a, bt, c);
            }
            store_c16x16_f32(Ps + mt * 16 * PS_LD + nt * 16, PS_LD, c);
        } else {
            // GEMM2: Yinter(32x32) = Qc @ S0 via bf16 shadow (pure b128 loads)
            int w = wave - 4;
            int mt = w >> 1, nt = w & 1;
            v8f c;
#pragma unroll
            for (int i = 0; i < 8; ++i) c[i] = 0.0f;
            for (int ks = 0; ks < DKn; ks += 32) {
                v16bf a  = load_a16x32_bf(Qs + mt * 16 * QS_LD + ks, QS_LD);
                v16bf bb = load_b32x16_nmajor_bf(Stb + nt * 16 * STB_LD + ks, STB_LD);
                c = wmma_bf16(a, bb, c);
            }
            store_c16x16_f32(Ys + mt * 16 * YS_LD + nt * 16, YS_LD, c);
        }
        __syncthreads();

        if (wave < 4) {
            // GEMM3: y = G .* (Yinter + (P.*causal) @ K2)
            int mt = wave >> 1, nt = wave & 1;
            v8f c = load_c16x16_f32(Ys + mt * 16 * YS_LD + nt * 16, YS_LD);
            v16bf a  = load_a16x32_masked_f32(Ps + mt * 16 * PS_LD, PS_LD, mt * 16);
            v16bf bb = load_b32x16_nmajor_bf(K2t + nt * 16 * K2T_LD, K2T_LD);
            c = wmma_bf16(a, bb, c);
            int nl = lane & 15, mb = (lane < 16) ? 0 : 8;
#pragma unroll
            for (int i = 0; i < 8; ++i) {
                int t = mt * 16 + mb + i;
                int kc = nt * 16 + nl;
                float val = c[i] * Gs[t * GS_LD + kc];
                yg[(size_t)(b * Tn + t0 + t) * Dn + h * DKn + kb * KBW + kc] = val;
            }
        }
        // GEMM4: St <- Gend .* (St + K2^T @ Vc); refresh bf16 shadow
        for (int tile = wave; tile < 64; tile += 8) {
            int mtc = tile & 1;        // kcol tile 0..1
            int ntd = tile >> 1;       // dv tile 0..31
            v8f c = load_c16x16_f32(St + mtc * 16 * ST_LD + ntd * 16, ST_LD);
            v16bf a  = load_a16x32_bf(K2t + mtc * 16 * K2T_LD, K2T_LD);
            v16bf bb = load_b32x16_nmajor_bf(Vst + ntd * 16 * VST_LD, VST_LD);
            c = wmma_bf16(a, bb, c);
            int nl = lane & 15;
            int mb = (lane < 16) ? 0 : 8;
#pragma unroll
            for (int i = 0; i < 8; ++i)
                c[i] *= Gs[(CC - 1) * GS_LD + mtc * 16 + mb + i];
            store_c16x16_f32(St + mtc * 16 * ST_LD + ntd * 16, ST_LD, c);
#pragma unroll
            for (int i = 0; i < 8; ++i)
                Stb[(mtc * 16 + mb + i) * STB_LD + ntd * 16 + nl] = (__bf16)c[i];
        }
        __syncthreads();
    }
}

// ---------------------------------------------------------------------------
// Launch
// ---------------------------------------------------------------------------
extern "C" void kernel_launch(void* const* d_in, const int* in_sizes, int n_in,
                              void* d_out, int out_size, void* d_ws, size_t ws_size,
                              hipStream_t stream) {
    (void)in_sizes; (void)n_in; (void)out_size; (void)ws_size;
    const float* x    = (const float*)d_in[0];
    const float* Wq   = (const float*)d_in[1];
    const float* Wk   = (const float*)d_in[2];
    const float* Wv   = (const float*)d_in[3];
    const float* Wo   = (const float*)d_in[4];
    const float* Wgam = (const float*)d_in[5];
    const float* bgam = (const float*)d_in[6];
    const float* Wlam = (const float*)d_in[7];
    const float* blam = (const float*)d_in[8];
    float* out = (float*)d_out;

    // workspace: q(bf16 32MiB) | v(bf16 32MiB) | kn(f32 64MiB) | y(f32 64MiB)
    char* ws = (char*)d_ws;
    const size_t NE = (size_t)Bn * Tn * Dn;          // 16,777,216 elements
    __bf16* qb = (__bf16*)ws;
    __bf16* vb = (__bf16*)(ws + NE * 2);
    float*  kn = (float*)(ws + NE * 4);
    float*  y  = (float*)(ws + NE * 8);

    dim3 gGrid(MROWS / 128, Dn / 64);
    dim3 blk(256);

    gemm_xwT_kernel<2><<<gGrid, blk, 0, stream>>>(x, Wq, qb);   // bf16 head-major
    gemm_xwT_kernel<1><<<gGrid, blk, 0, stream>>>(x, Wk, kn);   // f32  head-major
    gemm_xwT_kernel<2><<<gGrid, blk, 0, stream>>>(x, Wv, vb);   // bf16 head-major

    knorm_kernel<<<dim3(Bn * Hn * Tn), blk, 0, stream>>>(kn);

    gla_scan_kernel<<<dim3(Bn * Hn * NKB), blk, SCAN_LDS_BYTES, stream>>>(
        qb, kn, vb, y, Wgam, bgam, Wlam, blam);

    gemm_xwT_kernel<0><<<gGrid, blk, 0, stream>>>(y, Wo, out);  // f32 standard
}
